// WaveFunctionElectronUpDown_44513041056432
// MI455X (gfx1250) — matrix operand order repaired and checked by reading, not compiled, not run on hardware
//
#include <hip/hip_runtime.h>
#include <hip/hip_bf16.h>
#include <stdint.h>

// ---------------------------------------------------------------------------
// Problem constants (from the reference)
// ---------------------------------------------------------------------------
#define NSITES   64
#define DS_DIM   128   // 2 * NSITES
#define H_DIM    512
#define B_BATCH  2048
#define NUP      32
#define NDN      32
#define BM       64    // batch rows per workgroup tile

typedef __attribute__((ext_vector_type(16))) __bf16 bf16x16;
typedef __attribute__((ext_vector_type(8)))  float  floatx8;
typedef __attribute__((ext_vector_type(4)))  int    int4v;

// silu via single-cycle v_rcp_f32 (1 ulp) instead of the precise-division
// macro sequence (div_scale/rcp/fma-Newton/div_fixup): epilogue is 2 VALU ops
// per element instead of ~8.  Accuracy is far beyond bf16 matmul noise.
__device__ __forceinline__ float silu_f(float v) {
    return v * __builtin_amdgcn_rcpf(1.0f + __expf(-v));
}

// ---------------------------------------------------------------------------
// Async global->LDS copy (CDNA5 GLOBAL_LOAD_ASYNC_TO_LDS_B128, ASYNCcnt)
// builtin signature (from clang diagnostic): (int4v AS1* src, int4v AS3* dst,
// imm offset, imm cpol)
// ---------------------------------------------------------------------------
__device__ __forceinline__ void async_copy16(const __bf16* g, __bf16* l) {
#if __has_builtin(__builtin_amdgcn_global_load_async_to_lds_b128)
    int4v* gg = (int4v*)g;   // C-style cast drops const
    int4v* ll = (int4v*)l;
    __builtin_amdgcn_global_load_async_to_lds_b128(
        (__attribute__((address_space(1))) int4v*)gg,
        (__attribute__((address_space(3))) int4v*)ll, 0, 0);
#else
    *reinterpret_cast<uint4*>(l) = *reinterpret_cast<const uint4*>(g);
#endif
}

__device__ __forceinline__ void async_wait0() {
#if __has_builtin(__builtin_amdgcn_s_wait_asynccnt)
    __builtin_amdgcn_s_wait_asynccnt(0);
#else
    asm volatile("s_wait_asynccnt 0" ::: "memory");
#endif
}

// ---------------------------------------------------------------------------
// WMMA fragment helpers (CDNA5 16x16x32 bf16, wave32)
//
// A (16x32, row-major in LDS):  lanes 0-15 hold row M=lane with K runs
// {0..7} and {16..23}; lanes 16-31 hold row M=lane-16 with K runs {8..15}
// and {24..31}.  Each run is 8 bf16 = one 16B LDS load.
// ---------------------------------------------------------------------------
__device__ __forceinline__ bf16x16 load_a_frag(const __bf16* act, int width,
                                               int rowBase, int kBase, int lane) {
    union { bf16x16 v; uint4 q[2]; } f;
    const int row = rowBase + (lane & 15);
    const int c1  = kBase + ((lane < 16) ? 0 : 8);
    f.q[0] = *reinterpret_cast<const uint4*>(act + (size_t)row * width + c1);
    f.q[1] = *reinterpret_cast<const uint4*>(act + (size_t)row * width + c1 + 16);
    return f.v;
}

// B (32x16, K x N block pre-packed by pack_wmmaB): each 32x16 block is stored
// as 32 lanes x 16 bf16, so lane `l` loads 32 contiguous bytes (coalesced).
__device__ __forceinline__ bf16x16 load_b_frag(const __bf16* wp, long tileIdx, int lane) {
    union { bf16x16 v; uint4 q[2]; } f;
    const uint4* p = reinterpret_cast<const uint4*>(wp + tileIdx * 512 + lane * 16);
    f.q[0] = p[0];
    f.q[1] = p[1];
    return f.v;
}

__device__ __forceinline__ floatx8 wmma_bf16(bf16x16 a, bf16x16 b, floatx8 c) {
    // 8 args: (neg_a, A, neg_b, B, c_mod, C, reuse_a, reuse_b)
    return __builtin_amdgcn_wmma_f32_16x16x32_bf16(false, a, false, b,
                                                   (short)0, c, false, false);
}

// ---------------------------------------------------------------------------
// Weight packing: f32 [sites, K, N] row-major  ->  bf16 WMMA-B fragment order
// dst layout: [site][K/32][N/16][lane(32)][j(16)]
//   lane<16 : k = kt*32 + (j<8 ? j : 16+j-8),     n = lane
//   lane>=16: k = kt*32 + (j<8 ? 8+j : 24+j-8),   n = lane-16
// ---------------------------------------------------------------------------
__global__ void pack_wmmaB(const float* __restrict__ src, __bf16* __restrict__ dst,
                           int sites, int K, int N) {
    const long total = (long)sites * K * N;
    for (long i = (long)blockIdx.x * blockDim.x + threadIdx.x; i < total;
         i += (long)gridDim.x * blockDim.x) {
        long t = i;
        const int j  = (int)(t & 15);  t >>= 4;
        const int l  = (int)(t & 31);  t >>= 5;
        const int nt = (int)(t % (N / 16)); t /= (N / 16);
        const int kt = (int)(t % (K / 32)); t /= (K / 32);
        const int s  = (int)t;
        const int n    = l & 15;
        const int koff = ((l < 16) ? 0 : 8) + (j & 7) + ((j & 8) ? 16 : 0);
        const float v = src[((long)s * K + kt * 32 + koff) * N + nt * 16 + n];
        dst[i] = (__bf16)v;
    }
}

// int {0,1} occupancy -> bf16 feature vector, once, so GEMM kernels can
// async-DMA their activation tiles straight into LDS.
__global__ void xf_convert(const int* __restrict__ x, __bf16* __restrict__ xf) {
    const int i = blockIdx.x * blockDim.x + threadIdx.x;
    if (i < B_BATCH * DS_DIM)
        xf[i] = (__bf16)(float)x[i];
}

// ---------------------------------------------------------------------------
// Fused per-site amplitude MLP:
//   h1 = silu(xf @ W1[s] + b1[s])          [BM x 512]   (WMMA bf16)
//   h2 = silu(h1 @ W2[s] + b2[s])          [BM x 512]   (WMMA bf16)
//   logits = h2 @ W3[s] + b3[s]            [BM x 4]     (folded, ds_add_f32)
// grid = (NSITES, B/BM), block = 256 (8 wave32s)
// Loop nest is (ct -> kt -> 4x rt accumulators): each weight fragment is
// fetched exactly once per wave; per workgroup layer-2 reads exactly the
// 512 KB packed site weights (hits L2: all packed weights = 42 MB < 192 MB).
// ---------------------------------------------------------------------------
__global__ __launch_bounds__(256) void amp_sites_kernel(
    const __bf16* __restrict__ xf,    // [B, 128] bf16
    const __bf16* __restrict__ W1p,   // packed [S][4][32][512]
    const __bf16* __restrict__ W2p,   // packed [S][16][32][512]
    const float* __restrict__ W3,     // [S, 512, 4]
    const float* __restrict__ b1,     // [S, 512]
    const float* __restrict__ b2,     // [S, 512]
    const float* __restrict__ b3,     // [S, 4]
    float*       __restrict__ logits) // [B, S, 4]
{
    __shared__ __bf16 actA[BM * DS_DIM];   // 16 KB
    __shared__ __bf16 h1[BM * H_DIM];      // 64 KB
    __shared__ float  bias[H_DIM];         //  2 KB
    __shared__ float  logitAcc[BM * 4];    //  1 KB

    const int s       = blockIdx.x;
    const int rowBase = blockIdx.y * BM;
    const int tid     = threadIdx.x;
    const int lane    = tid & 31;
    const int wave    = tid >> 5;

    // async-DMA the xf tile into LDS (16 KB, 128 B128 transfers per wave)
    {
        const __bf16* xsrc = xf + (size_t)rowBase * DS_DIM;
        for (int i = tid; i < (BM * DS_DIM) / 8; i += 256)
            async_copy16(xsrc + (size_t)i * 8, actA + (size_t)i * 8);
    }
    for (int i = tid; i < H_DIM; i += 256)
        bias[i] = b1[(size_t)s * H_DIM + i];
    if (tid < BM * 4) logitAcc[tid] = 0.0f;
    async_wait0();
    __syncthreads();

    // ---- layer 1: [BM x 128] x [128 x 512] ----
    for (int ci = 0; ci < 4; ++ci) {
        const int ct = wave + ci * 8;
        floatx8 acc[4] = {};
        for (int kt = 0; kt < DS_DIM / 32; ++kt) {
            const long tile = ((long)(s * (DS_DIM / 32) + kt)) * (H_DIM / 16) + ct;
            bf16x16 bw = load_b_frag(W1p, tile, lane);
#pragma unroll
            for (int rt = 0; rt < 4; ++rt) {
                bf16x16 a = load_a_frag(actA, DS_DIM, rt * 16, kt * 32, lane);
                acc[rt] = wmma_bf16(a, bw, acc[rt]);
            }
        }
        const int   col = ct * 16 + (lane & 15);
        const float bc  = bias[col];
#pragma unroll
        for (int rt = 0; rt < 4; ++rt) {
#pragma unroll
            for (int r = 0; r < 8; ++r) {
                const int row = rt * 16 + ((lane < 16) ? r : 8 + r);
                h1[(size_t)row * H_DIM + col] = (__bf16)silu_f(acc[rt][r] + bc);
            }
        }
    }
    __syncthreads();
    for (int i = tid; i < H_DIM; i += 256)
        bias[i] = b2[(size_t)s * H_DIM + i];
    __syncthreads();

    // ---- layer 2 (+ fused layer 3 reduction): [BM x 512] x [512 x 512] ----
    for (int ci = 0; ci < 4; ++ci) {
        const int ct = wave + ci * 8;
        floatx8 acc[4] = {};
        for (int kt = 0; kt < H_DIM / 32; ++kt) {
            const long tile = ((long)(s * (H_DIM / 32) + kt)) * (H_DIM / 16) + ct;
            if (kt + 1 < H_DIM / 32)   // stream next K-slice of weights
                __builtin_prefetch(W2p + (tile + (H_DIM / 16)) * 512 + lane * 16, 0, 1);
            bf16x16 bw = load_b_frag(W2p, tile, lane);
#pragma unroll
            for (int rt = 0; rt < 4; ++rt) {
                bf16x16 a = load_a_frag(h1, H_DIM, rt * 16, kt * 32, lane);
                acc[rt] = wmma_bf16(a, bw, acc[rt]);
            }
        }
        const int   col = ct * 16 + (lane & 15);
        const float bc  = bias[col];
        const float4 w3v = *reinterpret_cast<const float4*>(W3 + ((size_t)s * H_DIM + col) * 4);
#pragma unroll
        for (int rt = 0; rt < 4; ++rt) {
#pragma unroll
            for (int r = 0; r < 8; ++r) {
                const int row = rt * 16 + ((lane < 16) ? r : 8 + r);
                const float v = silu_f(acc[rt][r] + bc);
                atomicAdd(&logitAcc[row * 4 + 0], v * w3v.x);
                atomicAdd(&logitAcc[row * 4 + 1], v * w3v.y);
                atomicAdd(&logitAcc[row * 4 + 2], v * w3v.z);
                atomicAdd(&logitAcc[row * 4 + 3], v * w3v.w);
            }
        }
    }
    __syncthreads();
    if (tid < BM * 4) {
        const int row = tid >> 2, o = tid & 3;
        logits[(((size_t)(rowBase + row)) * NSITES + s) * 4 + o] =
            logitAcc[tid] + b3[s * 4 + o];
    }
}

// ---------------------------------------------------------------------------
// Fused phase MLP: silu(xf@Wp1+bp1) -> silu(@Wp2+bp2) -> @Wp3+bp3 -> [B]
// grid = (B/BM), block = 256
// ---------------------------------------------------------------------------
__global__ __launch_bounds__(256) void phase_kernel(
    const __bf16* __restrict__ xf,    // [B, 128] bf16
    const __bf16* __restrict__ Wp1p,  // packed [4][32][512]
    const __bf16* __restrict__ Wp2p,  // packed [16][32][512]
    const float* __restrict__ Wp3,    // [512]
    const float* __restrict__ bp1,    // [512]
    const float* __restrict__ bp2,    // [512]
    const float* __restrict__ bp3,    // [1]
    float*       __restrict__ phase)  // [B]
{
    __shared__ __bf16 actA[BM * DS_DIM];
    __shared__ __bf16 h1[BM * H_DIM];
    __shared__ float  bias[H_DIM];
    __shared__ float  phAcc[BM];

    const int rowBase = blockIdx.x * BM;
    const int tid     = threadIdx.x;
    const int lane    = tid & 31;
    const int wave    = tid >> 5;

    {
        const __bf16* xsrc = xf + (size_t)rowBase * DS_DIM;
        for (int i = tid; i < (BM * DS_DIM) / 8; i += 256)
            async_copy16(xsrc + (size_t)i * 8, actA + (size_t)i * 8);
    }
    for (int i = tid; i < H_DIM; i += 256)
        bias[i] = bp1[i];
    if (tid < BM) phAcc[tid] = 0.0f;
    async_wait0();
    __syncthreads();

    // layer 1
    for (int ci = 0; ci < 4; ++ci) {
        const int ct = wave + ci * 8;
        floatx8 acc[4] = {};
        for (int kt = 0; kt < DS_DIM / 32; ++kt) {
            const long tile = (long)kt * (H_DIM / 16) + ct;
            bf16x16 bw = load_b_frag(Wp1p, tile, lane);
#pragma unroll
            for (int rt = 0; rt < 4; ++rt) {
                bf16x16 a = load_a_frag(actA, DS_DIM, rt * 16, kt * 32, lane);
                acc[rt] = wmma_bf16(a, bw, acc[rt]);
            }
        }
        const int   col = ct * 16 + (lane & 15);
        const float bc  = bias[col];
#pragma unroll
        for (int rt = 0; rt < 4; ++rt) {
#pragma unroll
            for (int r = 0; r < 8; ++r) {
                const int row = rt * 16 + ((lane < 16) ? r : 8 + r);
                h1[(size_t)row * H_DIM + col] = (__bf16)silu_f(acc[rt][r] + bc);
            }
        }
    }
    __syncthreads();
    for (int i = tid; i < H_DIM; i += 256)
        bias[i] = bp2[i];
    __syncthreads();

    // layer 2 + fused final dot with Wp3
    for (int ci = 0; ci < 4; ++ci) {
        const int ct = wave + ci * 8;
        floatx8 acc[4] = {};
        for (int kt = 0; kt < H_DIM / 32; ++kt) {
            const long tile = (long)kt * (H_DIM / 16) + ct;
            bf16x16 bw = load_b_frag(Wp2p, tile, lane);
#pragma unroll
            for (int rt = 0; rt < 4; ++rt) {
                bf16x16 a = load_a_frag(h1, H_DIM, rt * 16, kt * 32, lane);
                acc[rt] = wmma_bf16(a, bw, acc[rt]);
            }
        }
        const int   col = ct * 16 + (lane & 15);
        const float bc  = bias[col];
        const float w3  = Wp3[col];
#pragma unroll
        for (int rt = 0; rt < 4; ++rt) {
#pragma unroll
            for (int r = 0; r < 8; ++r) {
                const int row = rt * 16 + ((lane < 16) ? r : 8 + r);
                atomicAdd(&phAcc[row], silu_f(acc[rt][r] + bc) * w3);
            }
        }
    }
    __syncthreads();
    if (tid < BM)
        phase[rowBase + tid] = phAcc[tid] + bp3[0];
}

// ---------------------------------------------------------------------------
// Finalize: autoregressive mask + 0.5*logsumexp(2x) + gather + complex exp
// ---------------------------------------------------------------------------
__global__ void finalize_kernel(const int* __restrict__ x,
                                const float* __restrict__ logits,
                                const float* __restrict__ phase,
                                float* __restrict__ out) {
    const int b = blockIdx.x * blockDim.x + threadIdx.x;
    if (b >= B_BATCH) return;
    int up = 0, dn = 0;
    float amp = 0.0f;
    for (int s = 0; s < NSITES; ++s) {
        const int xu = x[((size_t)b * NSITES + s) * 2 + 0];
        const int xd = x[((size_t)b * NSITES + s) * 2 + 1];
        const bool au0 = (s - up) < (NSITES - NUP);
        const bool au1 = up < NUP;
        const bool ad0 = (s - dn) < (NSITES - NDN);
        const bool ad1 = dn < NDN;
        const bool m[4] = { au0 && ad0, au0 && ad1, au1 && ad0, au1 && ad1 };
        const float* lg = &logits[((size_t)b * NSITES + s) * 4];
        float d[4], mx = -INFINITY;
#pragma unroll
        for (int k = 0; k < 4; ++k) {
            d[k] = m[k] ? lg[k] : -INFINITY;
            mx = fmaxf(mx, 2.0f * d[k]);
        }
        float ssum = 0.0f;
#pragma unroll
        for (int k = 0; k < 4; ++k)
            ssum += __expf(2.0f * d[k] - mx);
        const float lse = 0.5f * (mx + __logf(ssum));
        const int idx = xu * 2 + xd;
        amp += d[idx] - lse;
        up += xu;
        dn += xd;
    }
    const float ph = phase[b];
    const float ea = __expf(amp);
    out[2 * b + 0] = ea * __cosf(ph);
    out[2 * b + 1] = ea * __sinf(ph);
}

// ---------------------------------------------------------------------------
// Host launcher
// ---------------------------------------------------------------------------
extern "C" void kernel_launch(void* const* d_in, const int* in_sizes, int n_in,
                              void* d_out, int out_size, void* d_ws, size_t ws_size,
                              hipStream_t stream) {
    (void)in_sizes; (void)n_in; (void)out_size; (void)ws_size;

    const int*   x   = (const int*)  d_in[0];
    const float* W1  = (const float*)d_in[1];
    const float* b1  = (const float*)d_in[2];
    const float* W2  = (const float*)d_in[3];
    const float* b2  = (const float*)d_in[4];
    const float* W3  = (const float*)d_in[5];
    const float* b3  = (const float*)d_in[6];
    const float* Wp1 = (const float*)d_in[7];
    const float* bp1 = (const float*)d_in[8];
    const float* Wp2 = (const float*)d_in[9];
    const float* bp2 = (const float*)d_in[10];
    const float* Wp3 = (const float*)d_in[11];
    const float* bp3 = (const float*)d_in[12];
    float* out = (float*)d_out;

    // workspace layout (bytes)
    char* ws = (char*)d_ws;
    const size_t W1P_BYTES    = (size_t)NSITES * DS_DIM * H_DIM * 2; //  8.4 MB
    const size_t W2P_BYTES    = (size_t)NSITES * H_DIM  * H_DIM * 2; // 33.6 MB
    const size_t WP1P_BYTES   = (size_t)DS_DIM * H_DIM * 2;          // 128 KB
    const size_t WP2P_BYTES   = (size_t)H_DIM  * H_DIM * 2;          // 512 KB
    const size_t LOGITS_BYTES = (size_t)B_BATCH * NSITES * 4 * 4;    //  2 MB
    const size_t PHASE_BYTES  = (size_t)B_BATCH * 4;                 //  8 KB

    size_t off = 0;
    __bf16* W1p   = (__bf16*)(ws + off); off += W1P_BYTES;
    __bf16* W2p   = (__bf16*)(ws + off); off += W2P_BYTES;
    __bf16* Wp1p  = (__bf16*)(ws + off); off += WP1P_BYTES;
    __bf16* Wp2p  = (__bf16*)(ws + off); off += WP2P_BYTES;
    float*  logit = (float*) (ws + off); off += LOGITS_BYTES;
    float*  phase = (float*) (ws + off); off += PHASE_BYTES;
    __bf16* xf    = (__bf16*)(ws + off);

    // pack weights into WMMA B-fragment order (bf16) + convert occupancies
    hipLaunchKernelGGL(pack_wmmaB, dim3(2048), dim3(256), 0, stream, W1,  W1p,  NSITES, DS_DIM, H_DIM);
    hipLaunchKernelGGL(pack_wmmaB, dim3(4096), dim3(256), 0, stream, W2,  W2p,  NSITES, H_DIM,  H_DIM);
    hipLaunchKernelGGL(pack_wmmaB, dim3(64),   dim3(256), 0, stream, Wp1, Wp1p, 1,      DS_DIM, H_DIM);
    hipLaunchKernelGGL(pack_wmmaB, dim3(256),  dim3(256), 0, stream, Wp2, Wp2p, 1,      H_DIM,  H_DIM);
    hipLaunchKernelGGL(xf_convert, dim3((B_BATCH * DS_DIM) / 256), dim3(256), 0, stream, x, xf);

    // fused amplitude network (64 sites x 32 batch tiles)
    hipLaunchKernelGGL(amp_sites_kernel, dim3(NSITES, B_BATCH / BM), dim3(256), 0, stream,
                       xf, W1p, W2p, W3, b1, b2, b3, logit);

    // fused phase network
    hipLaunchKernelGGL(phase_kernel, dim3(B_BATCH / BM), dim3(256), 0, stream,
                       xf, Wp1p, Wp2p, Wp3, bp1, bp2, bp3, phase);

    // mask/logsumexp/gather + complex exp
    hipLaunchKernelGGL(finalize_kernel, dim3((B_BATCH + 255) / 256), dim3(256), 0, stream,
                       x, logit, phase, out);
}
